// WindowAttention_67637144977770
// MI455X (gfx1250) — matrix operand order repaired and checked by reading, not compiled, not run on hardware
//
#include <hip/hip_runtime.h>

typedef __attribute__((ext_vector_type(16))) _Float16 v16h;
typedef __attribute__((ext_vector_type(8)))  _Float16 v8h;
typedef __attribute__((ext_vector_type(8)))  float    v8f;

#define WS 7
#define N_TOK 49
#define NPAD 64
#define NH 16
#define HD 32
#define ATT_SCALE 0.17677669529663687f   // 32^-0.5

#define WAVES_PER_BLOCK 2
// per-wave LDS layout (bytes):
//   q   : 64*32*2 = 4096  @ 0       (row-major, zero-padded rows 49..63)
//   k   : 4096            @ 4096
//   vt  : 32*64*2 = 4096  @ 8192    (V transposed: row = head channel d)
//   p   : 64*64*2 = 8192  @ 12288   (softmaxed probabilities, f16)
//   bias: 169*4 -> 704    @ 20480   (this head's bias column, f32)
//   mlut: 64*4   = 256    @ 21184   (packed per-row index constants)
#define WAVE_LDS 21504

__device__ __forceinline__ v16h load_a_frag(const _Float16* tile, int lane, int stride) {
  // 16-bit A-matrix 16x32 fragment (ISA 7.12.2): lane = M row; low half-wave
  // holds K 0-7 (v0-3) and 16-23 (v4-7), high half-wave K 8-15 and 24-31.
  union { v16h v; v8h h[2]; } u;
  const int m  = lane & 15;
  const int kb = (lane < 16) ? 0 : 8;
  const _Float16* rp = tile + m * stride;
  u.h[0] = *(const v8h*)(rp + kb);
  u.h[1] = *(const v8h*)(rp + kb + 16);
  return u.v;
}

__device__ __forceinline__ v16h load_b_frag(const _Float16* tile, int lane, int stride) {
  // 16-bit B-matrix 32x16 fragment: lane = N column; low half-wave holds
  // K 0-7 and 16-23, high half-wave K 8-15 and 24-31. `tile` holds 16
  // rows each being a K-contiguous run (K-matrix row / V^T row).
  union { v16h v; v8h h[2]; } u;
  const int n  = lane & 15;
  const int kb = (lane < 16) ? 0 : 8;
  const _Float16* rp = tile + n * stride + kb;
  u.h[0] = *(const v8h*)(rp);
  u.h[1] = *(const v8h*)(rp + 16);
  return u.v;
}

__global__ __launch_bounds__(WAVES_PER_BLOCK * 32)
void win_attn_kernel(const float* __restrict__ qkv,
                     const float* __restrict__ mask,
                     const float* __restrict__ bias_table,
                     float* __restrict__ out) {
  __shared__ __align__(16) unsigned char smem[WAVES_PER_BLOCK * WAVE_LDS];
  const int lane = threadIdx.x & 31;
  const int wid  = threadIdx.x >> 5;

  // Block-uniform (SGPR) window identity: 8 blocks per window, 2 heads/block.
  const int bs = blockIdx.x >> 3;            // window id 0..2047  (uniform)
  const int b  = bs >> 6;                    // image 0..31        (uniform)
  const int w  = bs & 63;                    // window-in-image    (uniform)
  const int wy = w >> 3, wx = w & 7;         // uniform
  const int h  = ((blockIdx.x & 7) << 1) + wid;  // head 0..15 (per-wave)

  unsigned char* base = smem + wid * WAVE_LDS;
  _Float16* qs  = (_Float16*)(base);
  _Float16* ks  = (_Float16*)(base + 4096);
  _Float16* vts = (_Float16*)(base + 8192);
  _Float16* ps  = (_Float16*)(base + 12288);
  float*    bf  = (float*)(base + 20480);
  int*      mlut = (int*)(base + 21184);

  const size_t plane = (size_t)32 * 56 * 56 * 512;
  const int ch = h * HD + lane;      // vector offset: head channels

  // ---- stage Q,K (row-major f16) and V^T into LDS, zero-pad to 64 ----
  for (int p = 0; p < N_TOK; ++p) {
    const int row = wy * WS + p / WS;                    // uniform
    const int col = wx * WS + p % WS;                    // uniform
    const size_t pix = (((size_t)b * 56 + row) * 56 + col) * 512;  // uniform
    const float* qp = qkv + pix;                         // SGPR base
    qs[p * HD + lane]    = (_Float16)qp[ch];
    ks[p * HD + lane]    = (_Float16)qp[plane + ch];
    vts[lane * NPAD + p] = (_Float16)qp[2 * plane + ch];
  }
  for (int p = N_TOK; p < NPAD; ++p) {
    qs[p * HD + lane] = (_Float16)0.f;
    ks[p * HD + lane] = (_Float16)0.f;
    vts[lane * NPAD + p] = (_Float16)0.f;
  }
  for (int t = lane; t < 169; t += 32) bf[t] = bias_table[t * NH + h];

  // per-row packed constants: low16 = 4*(mIdx), high16 = mask row offset.
  // mIdx = mcl + 6*(mcl/7) + 84 with mcl = min(m, 48).
  for (int t = lane; t < NPAD; t += 32) {
    const int mcl = (t < N_TOK) ? t : (N_TOK - 1);
    const int mIdxB = (mcl + 6 * (mcl / WS) + 84) << 2;   // byte offset, <= 672
    const int moff  = mcl * N_TOK;                        // <= 2352
    mlut[t] = mIdxB | (moff << 16);
  }

  // uniform (SGPR) mask base; index with 32-bit int vector offsets only
  const float* mrow = mask + w * (N_TOK * N_TOK);
  const int nl = lane & 15;
  const int hw = lane >> 4;
  const int* mlutL = mlut + 8 * hw;   // per-lane base; row index is immediate

  // Separable rel-pos index: idx(m,n) = (m + 6*(m/7)) - (n + 6*(n/7)) + 84.
  // Per-lane key-column constants (byte-scaled), computed once:
  int nIdxB[4];   // 4*(n + 6*(n/7)), clamped n
  int nOffJ[4];   // clamped n (mask column)
  int nokJ[4];
  #pragma unroll
  for (int j = 0; j < 4; ++j) {
    const int n  = 16 * j + nl;
    const int nc = (n < N_TOK - 1) ? n : (N_TOK - 1);   // clamp to 48
    nIdxB[j] = (nc + 6 * (nc / WS)) << 2;
    nOffJ[j] = nc;
    nokJ[j]  = (n < N_TOK);
  }
  const char* bfb = (const char*)bf;

  // ---- S = scale*Q K^T + bias + mask, softmax -> P (f16 in LDS) ----
  for (int i = 0; i < 4; ++i) {
    v16h aq = load_a_frag(qs + 16 * i * HD, lane, HD);
    v8f s[4];
    #pragma unroll
    for (int j = 0; j < 4; ++j) {
      v16h bk = load_b_frag(ks + 16 * j * HD, lane, HD);
      v8f z = {};
      s[j] = __builtin_amdgcn_wmma_f32_16x16x32_f16(false, aq, false, bk,
                                                    (short)0, z, false, false);
    }
    #pragma unroll
    for (int r = 0; r < 8; ++r) {
      const int m     = 16 * i + r + 8 * hw; // C-layout row
      const int mi    = mlutL[16 * i + r];   // ds_load, immediate offset
      const int mIdxB = mi & 0xffff;
      const int moff  = mi >> 16;
      const int mok   = (m < N_TOK);
      float mx = -1e30f;
      #pragma unroll
      for (int j = 0; j < 4; ++j) {
        // bias: single v_sub byte-offset ds_load; mask: saddr global load
        const float bias = *(const float*)(bfb + (mIdxB - nIdxB[j]));
        const float add  = bias + mrow[moff + nOffJ[j]];
        float val = s[j][r] * ATT_SCALE;
        val = mok     ? (val + add) : val;   // pad query rows: plain 0*scale
        val = nokJ[j] ? val : -1e30f;        // pad key cols: -inf
        s[j][r] = val;
        mx = fmaxf(mx, val);
      }
      #pragma unroll
      for (int x = 1; x < 16; x <<= 1) mx = fmaxf(mx, __shfl_xor(mx, x, 32));
      float sum = 0.f;
      #pragma unroll
      for (int j = 0; j < 4; ++j) {
        float e = __expf(s[j][r] - mx);
        s[j][r] = e;
        sum += e;
      }
      #pragma unroll
      for (int x = 1; x < 16; x <<= 1) sum += __shfl_xor(sum, x, 32);
      const float inv = 1.f / sum;
      #pragma unroll
      for (int j = 0; j < 4; ++j)
        ps[m * NPAD + 16 * j + nl] = (_Float16)(s[j][r] * inv);
    }
  }

  // ---- O = P V ----
  float* outw = out + (size_t)bs * N_TOK * 512;   // SGPR base (bs uniform)
  for (int i = 0; i < 4; ++i) {
    #pragma unroll
    for (int dt = 0; dt < 2; ++dt) {
      v8f acc = {};
      #pragma unroll
      for (int c = 0; c < 2; ++c) {
        v16h ap = load_a_frag(ps + 16 * i * NPAD + 32 * c, lane, NPAD);
        v16h bv = load_b_frag(vts + 16 * dt * NPAD + 32 * c, lane, NPAD);
        acc = __builtin_amdgcn_wmma_f32_16x16x32_f16(false, ap, false, bv,
                                                     (short)0, acc, false, false);
      }
      #pragma unroll
      for (int r = 0; r < 8; ++r) {
        const int m = 16 * i + r + 8 * hw;
        if (m < N_TOK) {
          outw[m * 512 + h * HD + 16 * dt + nl] = acc[r];
        }
      }
    }
  }
}

extern "C" void kernel_launch(void* const* d_in, const int* in_sizes, int n_in,
                              void* d_out, int out_size, void* d_ws, size_t ws_size,
                              hipStream_t stream) {
  (void)in_sizes; (void)n_in; (void)d_ws; (void)ws_size; (void)out_size;
  const float* qkv        = (const float*)d_in[0];
  const float* mask       = (const float*)d_in[1];
  const float* bias_table = (const float*)d_in[2];
  float* out = (float*)d_out;

  const int total_waves = 2048 * NH;                 // one wave per (window, head)
  const int blocks = total_waves / WAVES_PER_BLOCK;  // 16384
  win_attn_kernel<<<blocks, WAVES_PER_BLOCK * 32, 0, stream>>>(
      qkv, mask, bias_table, out);
}
